// NoisyRouter_47201690583343
// MI455X (gfx1250) — matrix operand order repaired and checked by reading, not compiled, not run on hardware
//
#include <hip/hip_runtime.h>
#include <hip/hip_bf16.h>
#include <math.h>
#include <stdint.h>

// Problem constants (from reference)
#define B_    4
#define S_    4096
#define DIM_  4096
#define E_    64
#define TOPK_ 8
#define NTOK  (B_ * S_)       // 16384 tokens
#define MT    128             // tokens per workgroup
#define KC    32              // K chunk (one bf16 WMMA k-step)
#define NC    (DIM_ / KC)     // 128 chunks
#define NCOL  128             // 64 gate cols + 64 noise cols

typedef __attribute__((ext_vector_type(16))) __bf16 v16bf;
typedef __attribute__((ext_vector_type(8)))  float  v8f;

// ---------------------------------------------------------------------------
// CDNA5 async global->LDS copy, 16 bytes per lane. Tracked by ASYNCcnt.
__device__ __forceinline__ void async_copy_b128(uint32_t lds_off, uint32_t voff,
                                                const void* base) {
  uint64_t b = (uint64_t)base;
  asm volatile("global_load_async_to_lds_b128 %0, %1, %2"
               :: "v"(lds_off), "v"(voff), "s"(b)
               : "memory");
}

__device__ __forceinline__ void wait_async0() {
  asm volatile("s_wait_asynccnt 0" ::: "memory");
}

// LDS byte address of a shared-memory pointer (flat LDS aperture: low 32 bits
// are the wave-relative LDS address).
__device__ __forceinline__ uint32_t lds_addr(const void* p) {
  return (uint32_t)(uintptr_t)p;
}

// ---------------------------------------------------------------------------
// Pre-pass: convert gate_w (64x4096) and noise_w (64x4096) f32 -> bf16 into
// d_ws, laid out as wbf[128][4096] (rows 0..63 gate, 64..127 noise).
__global__ __launch_bounds__(256)
void convert_weights_kernel(const float* __restrict__ gw,
                            const float* __restrict__ nw,
                            unsigned short* __restrict__ wraw) {
  __bf16* wbf = (__bf16*)wraw;
  int i = blockIdx.x * blockDim.x + threadIdx.x;   // 0 .. 131071
  int base = i * 4;
  const int half = E_ * DIM_;                      // 262144
  const float* src = (base < half) ? (gw + base) : (nw + (base - half));
  float4 v = *(const float4*)src;
  wbf[base + 0] = (__bf16)v.x;
  wbf[base + 1] = (__bf16)v.y;
  wbf[base + 2] = (__bf16)v.z;
  wbf[base + 3] = (__bf16)v.w;
}

// ---------------------------------------------------------------------------
// Issue async loads for one K-chunk:
//   x tile  [128 rows x 32 k] f32   -> 4 b128 per thread
//   w tile  [128 cols x 32 k] bf16  -> 2 b128 per thread
__device__ __forceinline__ void issue_chunk(const float*  __restrict__ x,
                                            const __bf16* __restrict__ wbf,
                                            int tok0, int c, int t,
                                            float (*lx)[KC], __bf16 (*lwb)[KC]) {
  const uint32_t k0 = (uint32_t)c * KC;
#pragma unroll
  for (int i = 0; i < 4; ++i) {           // 1024 x-segments of 4 floats
    int s  = t + 256 * i;
    int r  = s >> 3;                      // token row 0..127
    int ks = (s & 7) * 4;                 // k sub-offset 0..28
    uint32_t dst = lds_addr(&lx[r][ks]);
    uint32_t go  = (((uint32_t)(tok0 + r)) * DIM_ + k0 + ks) * 4u;
    async_copy_b128(dst, go, x);
  }
#pragma unroll
  for (int i = 0; i < 2; ++i) {           // 512 w-segments of 8 bf16
    int s  = t + 256 * i;
    int cc = s >> 2;                      // output column 0..127
    int ks = (s & 3) * 8;                 // k sub-offset 0,8,16,24
    uint32_t dst = lds_addr(&lwb[cc][ks]);
    uint32_t go  = ((uint32_t)cc * DIM_ + k0 + ks) * 2u;
    async_copy_b128(dst, go, wbf);
  }
}

// ---------------------------------------------------------------------------
__global__ __launch_bounds__(256, 1)
void noisy_router_kernel(const float* __restrict__ x,
                         const unsigned short* __restrict__ wraw,
                         const float* __restrict__ gb,
                         const float* __restrict__ nb,
                         const float* __restrict__ gauss,
                         float* __restrict__ out_probs,
                         int*   __restrict__ out_idx) {
  const __bf16* wbf = (const __bf16*)wraw;

  // 64 KB static LDS. During GEMM: [0,32K) = f32 x tiles (double-buffered),
  // [32K,48K) = bf16 w tiles (double-buffered). After GEMM the whole 64 KB is
  // re-used as the 128x128 f32 logits staging area (barrier-separated).
  __shared__ __align__(16) unsigned char smem[65536];
  float  (*lx)[MT][KC]    = (float  (*)[MT][KC])(smem);            // [2][128][32] f32
  __bf16 (*lwb)[NCOL][KC] = (__bf16 (*)[NCOL][KC])(smem + 32768);  // [2][128][32] bf16
  float  (*stage)[NCOL]   = (float  (*)[NCOL])(smem);              // [128][128] f32

  const int t    = threadIdx.x;
  const int wave = t >> 5;
  const int lane = t & 31;
  const int h    = lane >> 4;   // half-wave select
  const int l15  = lane & 15;
  const int tok0 = blockIdx.x * MT;

  v8f acc[8];
#pragma unroll
  for (int i = 0; i < 8; ++i)
#pragma unroll
    for (int j = 0; j < 8; ++j) acc[i][j] = 0.0f;

  // Prologue: fetch chunk 0
  issue_chunk(x, wbf, tok0, 0, t, lx[0], lwb[0]);
  wait_async0();
  __syncthreads();

  int buf = 0;
  for (int c = 0; c < NC; ++c) {
    if (c + 1 < NC)
      issue_chunk(x, wbf, tok0, c + 1, t, lx[buf ^ 1], lwb[buf ^ 1]);

    // Preload ALL eight B fragments into distinct registers.
    // B fragment: 32x16 bf16, col = nt*16 + l15, K = i + 16*h (32 contiguous B).
    v16bf bfrag[8];
#pragma unroll
    for (int nt = 0; nt < 8; ++nt) {
      const __bf16* wr = &lwb[buf][nt * 16 + l15][16 * h];
      bfrag[nt] = *(const v16bf*)wr;
    }

    // A fragment: 16x32 bf16, wave owns token rows [wave*16, wave*16+16).
    // Layout: lanes 0-15 M=lane, elems 0..15 -> K 0..15;
    // lanes 16-31 -> K 16..31 (matches B). Converted f32->bf16 in-register (8 pk-cvts).
    const int row = wave * 16 + l15;
    const float* xr = &lx[buf][row][0];
    v16bf a;
#pragma unroll
    for (int i = 0; i < 16; ++i) {
      int k = i + 16 * h;
      a[i] = (__bf16)xr[k];
    }

    // Scheduling fence: force all fragments to materialize (loads pipelined,
    // one dscnt drain) before any WMMA issues; prevents the scheduler from
    // sinking B-loads between WMMAs on a single recycled register pair.
#pragma unroll
    for (int nt = 0; nt < 8; ++nt)
      asm volatile("" : "+v"(bfrag[nt]));
    asm volatile("" : "+v"(a));

#pragma unroll
    for (int nt = 0; nt < 8; ++nt) {
      acc[nt] = __builtin_amdgcn_wmma_f32_16x16x32_bf16(
          false, a, false, bfrag[nt], (short)0, acc[nt], false, false);
    }

    wait_async0();      // next chunk's DMA has landed
    __syncthreads();    // all waves done with `buf` before it is re-filled
    buf ^= 1;
  }

  // Stage accumulators to LDS (aliases tile buffers; prior barrier protects).
  // C/D layout: elem j, lanes 0-15 -> M=j, N=lane; lanes 16-31 -> M=j+8.
#pragma unroll
  for (int nt = 0; nt < 8; ++nt)
#pragma unroll
    for (int j = 0; j < 8; ++j) {
      int M = wave * 16 + j + 8 * h;
      int N = nt * 16 + l15;
      stage[M][N] = acc[nt][j];
    }
  __syncthreads();

  // Epilogue: one thread per token. noisy = (gate+gb) + gauss*softplus(noise+nb)
  if (t < MT) {
    const int tok = tok0 + t;
    const float* grow = gauss + (size_t)tok * E_;
#pragma unroll 4
    for (int e = 0; e < E_; ++e) {
      float lg = stage[t][e] + gb[e];
      float nl = stage[t][E_ + e] + nb[e];
      float sp = (nl > 20.0f) ? nl : log1pf(__expf(nl));
      stage[t][e] = lg + grow[e] * sp;
    }

    // Serial top-8 (strict '>' => first-occurrence tie-break, like lax.top_k)
    unsigned long long mask = 0ull;
    float tv[TOPK_];
    int   ti[TOPK_];
    for (int kk = 0; kk < TOPK_; ++kk) {
      float best = -__builtin_inff();
      int   bi   = 0;
      for (int e = 0; e < E_; ++e) {
        float v = stage[t][e];
        if (((mask >> e) & 1ull) == 0ull && v > best) { best = v; bi = e; }
      }
      mask |= (1ull << bi);
      tv[kk] = best;
      ti[kk] = bi;
    }

    // Masked softmax: non-selected -> 0, selected -> softmax among the 8.
    float m = tv[0];
    float ev[TOPK_];
    float sum = 0.0f;
#pragma unroll
    for (int kk = 0; kk < TOPK_; ++kk) { ev[kk] = __expf(tv[kk] - m); sum += ev[kk]; }
    float inv = 1.0f / sum;

    float* orow = out_probs + (size_t)tok * E_;
#pragma unroll 4
    for (int e = 0; e < E_; ++e) orow[e] = 0.0f;
#pragma unroll
    for (int kk = 0; kk < TOPK_; ++kk) orow[ti[kk]] = ev[kk] * inv;

    int* irow = out_idx + (size_t)tok * TOPK_;
#pragma unroll
    for (int kk = 0; kk < TOPK_; ++kk) irow[kk] = ti[kk];
  }
}

// ---------------------------------------------------------------------------
extern "C" void kernel_launch(void* const* d_in, const int* in_sizes, int n_in,
                              void* d_out, int out_size, void* d_ws, size_t ws_size,
                              hipStream_t stream) {
  const float* x  = (const float*)d_in[0];
  const float* gw = (const float*)d_in[1];
  const float* gb = (const float*)d_in[2];
  const float* nw = (const float*)d_in[3];
  const float* nb = (const float*)d_in[4];
  const float* gs = (const float*)d_in[5];

  unsigned short* wbf = (unsigned short*)d_ws;   // [128][4096] bf16 (1 MB)

  float* probs = (float*)d_out;                               // [NTOK, E]
  int*   idx   = (int*)((float*)d_out + (size_t)NTOK * E_);   // [NTOK, TOPK]

  // Pre-pass: f32 weights -> bf16 (3 MB of traffic, one-time).
  convert_weights_kernel<<<dim3(512), dim3(256), 0, stream>>>(gw, nw, wbf);

  // Main fused GEMM + noisy-top-k + masked-softmax kernel.
  dim3 grid(NTOK / MT);   // 128 workgroups
  dim3 block(256);        // 8 wave32s
  noisy_router_kernel<<<grid, block, 0, stream>>>(x, wbf, gb, nb, gs, probs, idx);
}
